// SparseMoE_63419487093416
// MI455X (gfx1250) — matrix operand order repaired and checked by reading, not compiled, not run on hardware
//
#include <hip/hip_runtime.h>
#include <math.h>

// ---- problem constants (from reference) ----
#define Bb   8
#define Tt   2048
#define Cc   1024
#define Ee   8
#define Hh   4096
#define Nn   (Bb*Tt)        // 16384 tokens
#define CAPc 4096
#define LBW  0.01f

typedef __bf16 bf16;
typedef __attribute__((ext_vector_type(16))) __bf16 v16bf;
typedef __attribute__((ext_vector_type(8)))  float  v8f;

// ---- workspace layout (bytes) ----
#define OFF_GATES 0u
#define OFF_MASK  524288u
#define OFF_DISP  655360u
#define OFF_GSUM  786432u
#define OFF_H     1048576u   // CAP*H bf16 = 32MB, reused per expert

#define LDA 36               // bf16 elems per LDS row (72B: 8B-aligned, odd dword stride/2)
#define LDB 36

// ------------------------------------------------------------------
__global__ __launch_bounds__(256) void zero_out_kernel(float* __restrict__ out, long n) {
    long i = (long)blockIdx.x * blockDim.x + threadIdx.x;
    long stride = (long)gridDim.x * blockDim.x;
    for (; i < n; i += stride) out[i] = 0.0f;
}

__global__ __launch_bounds__(256) void init_misc_kernel(int* __restrict__ disp, float* __restrict__ gsum) {
    int i = blockIdx.x * blockDim.x + threadIdx.x;
    if (i < Ee * CAPc) disp[i] = Nn;
    if (i < Ee) gsum[i] = 0.0f;
}

// ------------------------------------------------------------------
// Router: one wave32 per token, 8 waves/block.
__global__ __launch_bounds__(256) void router_kernel(
    const float* __restrict__ x, const float* __restrict__ noise,
    const float* __restrict__ wr, const float* __restrict__ br,
    const float* __restrict__ wn, const float* __restrict__ bn,
    float* __restrict__ gates, unsigned char* __restrict__ mask,
    float* __restrict__ gsum)
{
    __shared__ float gsum_lds[Ee];
    if (threadIdx.x < Ee) gsum_lds[threadIdx.x] = 0.0f;
    __syncthreads();

    const int lane = threadIdx.x & 31;
    const int wid  = threadIdx.x >> 5;
    const int t    = blockIdx.x * 8 + wid;

    float aR[Ee], aN[Ee];
#pragma unroll
    for (int e = 0; e < Ee; ++e) { aR[e] = 0.f; aN[e] = 0.f; }

    const float* xt = x + (size_t)t * Cc;
    for (int c = lane; c < Cc; c += 32) {
        float xv = xt[c];
#pragma unroll
        for (int e = 0; e < Ee; ++e) {
            aR[e] = fmaf(xv, wr[e * Cc + c], aR[e]);
            aN[e] = fmaf(xv, wn[e * Cc + c], aN[e]);
        }
    }
#pragma unroll
    for (int e = 0; e < Ee; ++e) {
        for (int off = 16; off; off >>= 1) {
            aR[e] += __shfl_xor(aR[e], off, 32);
            aN[e] += __shfl_xor(aN[e], off, 32);
        }
    }

    if (lane == 0) {
        float nz[Ee];
#pragma unroll
        for (int e = 0; e < Ee; ++e) {
            float lg = aR[e] + br[e];
            float z  = aN[e] + bn[e];
            float sp = fmaxf(z, 0.f) + log1pf(expf(-fabsf(z)));
            nz[e] = lg + noise[(size_t)t * Ee + e] * sp;
        }
        int i1 = 0; float v1 = nz[0];
#pragma unroll
        for (int e = 1; e < Ee; ++e) if (nz[e] > v1) { v1 = nz[e]; i1 = e; }
        int i2 = -1; float v2 = -1e30f;
#pragma unroll
        for (int e = 0; e < Ee; ++e) if (e != i1 && nz[e] > v2) { v2 = nz[e]; i2 = e; }
        float e2 = expf(v2 - v1);
        float inv = 1.0f / (1.0f + e2);
        float g1 = inv, g2 = e2 * inv;
#pragma unroll
        for (int e = 0; e < Ee; ++e) {
            float g = (e == i1) ? g1 : (e == i2) ? g2 : 0.0f;
            gates[(size_t)t * Ee + e] = g;
            mask[(size_t)t * Ee + e] = (e == i1 || e == i2) ? 1 : 0;
        }
        atomicAdd(&gsum_lds[i1], g1);
        atomicAdd(&gsum_lds[i2], g2);
    }
    __syncthreads();
    if (threadIdx.x < Ee) atomicAdd(&gsum[threadIdx.x], gsum_lds[threadIdx.x]);
}

// ------------------------------------------------------------------
// Capacity scan: one wave per expert, sequential over token order.
__global__ __launch_bounds__(32) void scan_kernel(const unsigned char* __restrict__ mask,
                                                  int* __restrict__ disp)
{
    const int e = blockIdx.x;
    const int lane = threadIdx.x;
    int base = 0;
    for (int it = 0; it < Nn / 32; ++it) {
        int t = it * 32 + lane;
        int m = mask[(size_t)t * Ee + e];
        int v = m;
#pragma unroll
        for (int off = 1; off < 32; off <<= 1) {
            int u = __shfl_up(v, off, 32);
            if (lane >= off) v += u;
        }
        int pos = base + v - 1;
        if (m && pos < CAPc) disp[e * CAPc + pos] = t;
        base += __shfl(v, 31, 32);
    }
}

// ------------------------------------------------------------------
__device__ __forceinline__ unsigned pk_bf16(float a, float b) {
    union { unsigned u; bf16 h[2]; } t;
    t.h[0] = (bf16)a; t.h[1] = (bf16)b;
    return t.u;
}

// WMMA bf16 fragment loaders (CDNA5 ISA §7.12.2, wave32)
// A 16x32: interleaved-half K mapping; pairs contiguous -> dword LDS loads
__device__ __forceinline__ v16bf frag_A(const bf16* __restrict__ rowp, int half) {
    v16bf f;
#pragma unroll
    for (int j = 0; j < 8; ++j) {
        int k = ((j & 3) * 2) + ((j >> 2) * 16) + half * 8;
        f[2 * j]     = rowp[k];
        f[2 * j + 1] = rowp[k + 1];
    }
    return f;
}
// B 32x16 from transposed LDS tile Bt[n][k]: interleaved-half K mapping, pairs contiguous
__device__ __forceinline__ v16bf frag_B(const bf16* __restrict__ rowp, int half) {
    v16bf f;
#pragma unroll
    for (int j = 0; j < 8; ++j) {
        int k = ((j & 3) * 2) + ((j >> 2) * 16) + half * 8;
        f[2 * j]     = rowp[k];
        f[2 * j + 1] = rowp[k + 1];
    }
    return f;
}

// ------------------------------------------------------------------
// GEMM1: H[row,n] = relu( X[disp[row]] . W1[e] + b1[e] ) -> bf16 hbuf
// 128x128 tile, 8 waves (2x4), software-pipelined register staging.
__global__ __launch_bounds__(256) void gemm1_kernel(
    const float* __restrict__ x, const float* __restrict__ w1,
    const float* __restrict__ b1, const int* __restrict__ disp,
    bf16* __restrict__ hbuf, int e)
{
    __shared__ bf16 As[128 * LDA];      // [m][k]
    __shared__ bf16 Bt[128 * LDB];      // [n][k] (transposed)
    __shared__ int  tok[128];

    const int tid = threadIdx.x;
    const int mbase = blockIdx.y * 128;
    const int nbase = blockIdx.x * 128;
    if (tid < 128) tok[tid] = disp[e * CAPc + mbase + tid];
    __syncthreads();

    const int lane = tid & 31, wid = tid >> 5;
    const int waveM = wid >> 2, waveN = wid & 3;     // 2 x 4
    const int half = lane >> 4;

    v8f acc[4][2];
#pragma unroll
    for (int mi = 0; mi < 4; ++mi)
#pragma unroll
        for (int ni = 0; ni < 2; ++ni)
#pragma unroll
            for (int r = 0; r < 8; ++r) acc[mi][ni][r] = 0.0f;

    unsigned aS[8], bS[8];

#define G1_STAGE(KT) do {                                                         \
    _Pragma("unroll") for (int i = 0; i < 8; ++i) {                               \
        int idx = i * 256 + tid; int r = idx >> 4, kd = idx & 15;                 \
        int tkn = tok[r]; int tc = (tkn < Nn) ? tkn : 0;                          \
        const float* p = x + (size_t)tc * Cc + (KT) + 2 * kd;                     \
        float f0 = p[0], f1 = p[1];                                               \
        if (tkn >= Nn) { f0 = 0.f; f1 = 0.f; }                                    \
        aS[i] = pk_bf16(f0, f1); }                                                \
    _Pragma("unroll") for (int i = 0; i < 8; ++i) {                               \
        int idx = i * 256 + tid; int kd = idx >> 7, n = idx & 127;                \
        const float* p = w1 + ((size_t)e * Cc + (KT) + 2 * kd) * Hh + nbase + n;  \
        bS[i] = pk_bf16(p[0], p[Hh]); }                                           \
} while (0)

#define G1_COMMIT() do {                                                          \
    _Pragma("unroll") for (int i = 0; i < 8; ++i) {                               \
        int idx = i * 256 + tid; int r = idx >> 4, kd = idx & 15;                 \
        ((unsigned*)&As[r * LDA])[kd] = aS[i]; }                                  \
    _Pragma("unroll") for (int i = 0; i < 8; ++i) {                               \
        int idx = i * 256 + tid; int kd = idx >> 7, n = idx & 127;                \
        ((unsigned*)&Bt[n * LDB])[kd] = bS[i]; }                                  \
} while (0)

    G1_STAGE(0);
    for (int kt = 0; kt < Cc; kt += 32) {
        G1_COMMIT();
        __syncthreads();
        if (kt + 32 < Cc) G1_STAGE(kt + 32);    // overlap next-tile loads with WMMA

        v16bf a[4], b[2];
#pragma unroll
        for (int mi = 0; mi < 4; ++mi)
            a[mi] = frag_A(&As[(waveM * 64 + mi * 16 + (lane & 15)) * LDA], half);
#pragma unroll
        for (int ni = 0; ni < 2; ++ni)
            b[ni] = frag_B(&Bt[(waveN * 32 + ni * 16 + (lane & 15)) * LDB], half);
#pragma unroll
        for (int mi = 0; mi < 4; ++mi)
#pragma unroll
            for (int ni = 0; ni < 2; ++ni)
                acc[mi][ni] = __builtin_amdgcn_wmma_f32_16x16x32_bf16(
                    false, a[mi], false, b[ni], (short)0, acc[mi][ni], false, false);
        __syncthreads();
    }

#pragma unroll
    for (int ni = 0; ni < 2; ++ni) {
        int ncol = nbase + waveN * 32 + ni * 16 + (lane & 15);
        float bias = b1[e * Hh + ncol];
#pragma unroll
        for (int mi = 0; mi < 4; ++mi)
#pragma unroll
            for (int r = 0; r < 8; ++r) {
                int row = mbase + waveM * 64 + mi * 16 + half * 8 + r;
                float v = fmaxf(acc[mi][ni][r] + bias, 0.0f);
                hbuf[(size_t)row * Hh + ncol] = (bf16)v;
            }
    }
#undef G1_STAGE
#undef G1_COMMIT
}

// ------------------------------------------------------------------
// GEMM2: y = H . W2[e] + b2[e]; scatter-add y*gate into out (atomic f32).
// A tile (bf16 H) streamed with double-buffered global_load_async_to_lds_b64
// synchronized via s_wait_asynccnt; B tile register-staged f32->bf16.
__global__ __launch_bounds__(256) void gemm2_kernel(
    const bf16* __restrict__ hbuf, const float* __restrict__ w2,
    const float* __restrict__ b2, const int* __restrict__ disp,
    const float* __restrict__ gates, float* __restrict__ out, int e)
{
    __shared__ bf16 Abuf[2][128 * LDA];  // [m][k], double-buffered
    __shared__ bf16 Bt[128 * LDB];       // [n][k]
    __shared__ int  tok[128];

    const int tid = threadIdx.x;
    const int mbase = blockIdx.y * 128;
    const int nbase = blockIdx.x * 128;
    if (tid < 128) tok[tid] = disp[e * CAPc + mbase + tid];

    const int lane = tid & 31, wid = tid >> 5;
    const int waveM = wid >> 2, waveN = wid & 3;
    const int half = lane >> 4;

    v8f acc[4][2];
#pragma unroll
    for (int mi = 0; mi < 4; ++mi)
#pragma unroll
        for (int ni = 0; ni < 2; ++ni)
#pragma unroll
            for (int r = 0; r < 8; ++r) acc[mi][ni][r] = 0.0f;

    unsigned bS[8];

    // 4 async b64 instructions per thread per 128x32 bf16 tile (8KB)
#define G2_ASYNC(KT, BUF) do {                                                    \
    _Pragma("unroll") for (int i = 0; i < 4; ++i) {                               \
        int idx = i * 256 + tid; int r = idx >> 3, q = idx & 7;                   \
        const bf16* g = hbuf + (size_t)(mbase + r) * Hh + (KT) + q * 4;           \
        unsigned l = (unsigned)(unsigned long long)&Abuf[BUF][r * LDA + q * 4];   \
        asm volatile("global_load_async_to_lds_b64 %0, %1, off"                   \
                     :: "v"(l), "v"(g) : "memory"); }                             \
} while (0)

#define G2_STAGE_B(KT) do {                                                       \
    _Pragma("unroll") for (int i = 0; i < 8; ++i) {                               \
        int idx = i * 256 + tid; int kd = idx >> 7, n = idx & 127;                \
        const float* p = w2 + ((size_t)e * Hh + (KT) + 2 * kd) * Cc + nbase + n;  \
        bS[i] = pk_bf16(p[0], p[Cc]); }                                           \
} while (0)

#define G2_COMMIT_B() do {                                                        \
    _Pragma("unroll") for (int i = 0; i < 8; ++i) {                               \
        int idx = i * 256 + tid; int kd = idx >> 7, n = idx & 127;                \
        ((unsigned*)&Bt[n * LDB])[kd] = bS[i]; }                                  \
} while (0)

    G2_ASYNC(0, 0);
    G2_STAGE_B(0);
    const int Ks = Hh / 32;              // 128 K-steps
    for (int ki = 0; ki < Ks; ++ki) {
        const int cur = ki & 1;
        G2_COMMIT_B();
        if (ki + 1 < Ks) {
            G2_ASYNC((ki + 1) * 32, cur ^ 1);
            asm volatile("s_wait_asynccnt 0x4" ::: "memory");  // tile ki landed (in-order)
        } else {
            asm volatile("s_wait_asynccnt 0x0" ::: "memory");
        }
        __syncthreads();                 // all waves' async A + Bt stores visible
        if (ki + 1 < Ks) G2_STAGE_B((ki + 1) * 32);

        const bf16* Ac = Abuf[cur];
        v16bf a[4], b[2];
#pragma unroll
        for (int mi = 0; mi < 4; ++mi)
            a[mi] = frag_A(&Ac[(waveM * 64 + mi * 16 + (lane & 15)) * LDA], half);
#pragma unroll
        for (int ni = 0; ni < 2; ++ni)
            b[ni] = frag_B(&Bt[(waveN * 32 + ni * 16 + (lane & 15)) * LDB], half);
#pragma unroll
        for (int mi = 0; mi < 4; ++mi)
#pragma unroll
            for (int ni = 0; ni < 2; ++ni)
                acc[mi][ni] = __builtin_amdgcn_wmma_f32_16x16x32_bf16(
                    false, a[mi], false, b[ni], (short)0, acc[mi][ni], false, false);
        __syncthreads();
    }

    float bias[2];
#pragma unroll
    for (int ni = 0; ni < 2; ++ni)
        bias[ni] = b2[e * Cc + nbase + waveN * 32 + ni * 16 + (lane & 15)];

#pragma unroll
    for (int mi = 0; mi < 4; ++mi)
#pragma unroll
        for (int r = 0; r < 8; ++r) {
            int rloc = waveM * 64 + mi * 16 + half * 8 + r;
            int tkn = tok[rloc];
            if (tkn < Nn) {
                float g = gates[(size_t)tkn * Ee + e];
#pragma unroll
                for (int ni = 0; ni < 2; ++ni) {
                    int ncol = nbase + waveN * 32 + ni * 16 + (lane & 15);
                    float v = (acc[mi][ni][r] + bias[ni]) * g;
                    float* addr = out + (size_t)tkn * Cc + ncol;
                    asm volatile("global_atomic_add_f32 %0, %1, off"
                                 :: "v"(addr), "v"(v) : "memory");
                }
            }
        }
#undef G2_ASYNC
#undef G2_STAGE_B
#undef G2_COMMIT_B
}

// ------------------------------------------------------------------
__global__ void loss_kernel(const float* __restrict__ gsum, float* __restrict__ out) {
    if (threadIdx.x == 0 && blockIdx.x == 0) {
        float acc = 0.0f;
#pragma unroll
        for (int e = 0; e < Ee; ++e) {
            float frac = gsum[e] / (float)Nn;
            float d = frac - 1.0f / (float)Ee;
            acc += d * d;
        }
        out[(size_t)Nn * Cc] = LBW * (acc / (float)Ee);
    }
}

// ------------------------------------------------------------------
extern "C" void kernel_launch(void* const* d_in, const int* in_sizes, int n_in,
                              void* d_out, int out_size, void* d_ws, size_t ws_size,
                              hipStream_t stream) {
    const float* x       = (const float*)d_in[0];
    const float* noise   = (const float*)d_in[1];
    const float* w_route = (const float*)d_in[2];
    const float* b_route = (const float*)d_in[3];
    const float* w_noise = (const float*)d_in[4];
    const float* b_noise = (const float*)d_in[5];
    const float* w1      = (const float*)d_in[6];
    const float* b1      = (const float*)d_in[7];
    const float* w2      = (const float*)d_in[8];
    const float* b2      = (const float*)d_in[9];
    float* out = (float*)d_out;

    char* ws = (char*)d_ws;
    float*         gates = (float*)(ws + OFF_GATES);
    unsigned char* mask  = (unsigned char*)(ws + OFF_MASK);
    int*           disp  = (int*)(ws + OFF_DISP);
    float*         gsum  = (float*)(ws + OFF_GSUM);
    bf16*          hbuf  = (bf16*)(ws + OFF_H);

    zero_out_kernel<<<4096, 256, 0, stream>>>(out, (long)Nn * Cc);
    init_misc_kernel<<<(Ee * CAPc + 255) / 256, 256, 0, stream>>>(disp, gsum);
    router_kernel<<<Nn / 8, 256, 0, stream>>>(x, noise, w_route, b_route,
                                              w_noise, b_noise, gates, mask, gsum);
    scan_kernel<<<Ee, 32, 0, stream>>>(mask, disp);
    for (int e = 0; e < Ee; ++e) {
        gemm1_kernel<<<dim3(Hh / 128, CAPc / 128), 256, 0, stream>>>(x, w1, b1, disp, hbuf, e);
        gemm2_kernel<<<dim3(Cc / 128, CAPc / 128), 256, 0, stream>>>(hbuf, w2, b2, disp, gates, out, e);
    }
    loss_kernel<<<1, 1, 0, stream>>>(gsum, out);
}
